// MergedAttention_81131932221763
// MI455X (gfx1250) — matrix operand order repaired and checked by reading, not compiled, not run on hardware
//
#include <hip/hip_runtime.h>

// ---------------------------------------------------------------------------
// Merged two-stream attention == online softmax over concatenated KV (8192).
// S^T = K * Q^T   (keys x queries)  -> softmax per lane + one xor-16 shuffle
// O^T = V^T * P^T (d x queries)     -> V pre-transposed into workspace
// K / V^T chunks staged per-block into LDS via global_load_async_to_lds_b128
// (triple buffered, ASYNCcnt, one barrier/iter), operands via ds_load_b128.
// Register-pin asm forces batched ds_loads -> single wait -> wmma bursts.
// All GEMMs: v_wmma_f32_16x16x32_f16 (fp16 in, fp32 accumulate).
// ---------------------------------------------------------------------------

typedef __attribute__((ext_vector_type(16))) _Float16 v16h;
typedef __attribute__((ext_vector_type(8)))  _Float16 v8h;
typedef __attribute__((ext_vector_type(8)))  float    v8f;

#define B_   2
#define H_   16
#define Q_   1024
#define KV_  4096
#define D_   128
#define BH_  (B_ * H_)

#define KPAD 136   // 128 + 8 halves  -> 272B row stride (conflict-free b128)
#define VPAD 40    // 32 + 8 halves   -> 80B  row stride (conflict-free b128)
#define NBUF 3
#define NCHUNK ((2 * KV_) / 32)   // 256 chunks of 32 keys across both streams

union V16 { v16h v; v8h h[2]; int i[8]; };
union PK  { _Float16 f[2]; int i; };

// async 16B global -> LDS copy (one per lane), tracked by ASYNCcnt
#define ASYNC16(ldsptr, gptr)                                              \
    asm volatile("global_load_async_to_lds_b128 %0, %1, off"               \
                 :: "v"((unsigned)(size_t)(ldsptr)),                       \
                    "v"((unsigned long long)(size_t)(gptr))                \
                 : "memory")

static __device__ __forceinline__ void wait_async_le8() {
    asm volatile("s_wait_asynccnt 8" ::: "memory");
}
static __device__ __forceinline__ void wait_async_0() {
    asm volatile("s_wait_asynccnt 0" ::: "memory");
}

// --------------------------- V transpose pass ------------------------------
// VT[bh][d][kv] = V[bh][kv][d], 16x16 tiles via LDS (padded to kill conflicts)
__global__ __launch_bounds__(256) void vtrans_kernel(
    const _Float16* __restrict__ vA, const _Float16* __restrict__ vB,
    _Float16* __restrict__ vtA, _Float16* __restrict__ vtB)
{
    __shared__ _Float16 tile[16][17];
    const int z  = blockIdx.z;               // 0..63 : [stream | bh]
    const _Float16* __restrict__ src = (z & 32) ? vB  : vA;
    _Float16* __restrict__       dst = (z & 32) ? vtB : vtA;
    const int bh  = z & 31;
    const int kv0 = blockIdx.x * 16;
    const int d0  = blockIdx.y * 16;
    const int r   = threadIdx.x >> 4;
    const int c   = threadIdx.x & 15;

    tile[r][c] = src[((size_t)(bh * KV_ + kv0 + r)) * D_ + d0 + c];
    __syncthreads();
    dst[((size_t)(bh * D_ + d0 + r)) * KV_ + kv0 + c] = tile[c][r];
}

// --------------------------- flash attention -------------------------------
__global__ __launch_bounds__(128, 1) void mattn_kernel(
    const _Float16* __restrict__ q,
    const _Float16* __restrict__ kA,
    const _Float16* __restrict__ kB,
    const _Float16* __restrict__ vtA,
    const _Float16* __restrict__ vtB,
    _Float16* __restrict__ out,
    float* __restrict__ lse_out)
{
    __shared__ _Float16 Kl[NBUF][32][KPAD];    // 32 keys x 128 d  (26.1 KB)
    __shared__ _Float16 Vl[NBUF][128][VPAD];   // 128 d  x 32 keys (30.7 KB)

    const int tid   = threadIdx.x;
    const int lane  = tid & 31;
    const int wave  = tid >> 5;
    const int bh    = blockIdx.y;
    const int qbase = (blockIdx.x * 4 + wave) * 16;
    const int l16   = lane & 15;            // query column / key row / d row
    const int hi8   = (lane >> 4) * 8;      // half-wave element offset (A / C)
    const int hi16  = (lane >> 4) * 16;     // half-wave element offset (unused)
    const bool lo   = lane < 16;
    const float scale = 0.08838834764831845f;   // 1/sqrt(128)

    const _Float16* __restrict__ kAb = kA  + ((size_t)bh * KV_) * D_;
    const _Float16* __restrict__ kBb = kB  + ((size_t)bh * KV_) * D_;
    const _Float16* __restrict__ vAb = vtA + ((size_t)bh * D_) * (size_t)KV_;
    const _Float16* __restrict__ vBb = vtB + ((size_t)bh * D_) * (size_t)KV_;

    // cooperative async stage of one 32-key chunk (K: 8KB, V^T: 8KB) -> 8 ops
    auto stage = [&](int buf, int chunk) {
        const _Float16* kp = (chunk < NCHUNK / 2) ? kAb : kBb;
        const _Float16* vp = (chunk < NCHUNK / 2) ? vAb : vBb;
        const int kb = (chunk & (NCHUNK / 2 - 1)) * 32;
        #pragma unroll
        for (int i = 0; i < 4; ++i) {               // K: 512 x 16B transfers
            const int idx = i * 128 + tid;
            const int r = idx >> 4, c = idx & 15;
            ASYNC16(&Kl[buf][r][c * 8], kp + (size_t)(kb + r) * D_ + c * 8);
        }
        #pragma unroll
        for (int i = 0; i < 4; ++i) {               // V^T: 512 x 16B transfers
            const int idx = i * 128 + tid;
            const int r = idx >> 2, c = idx & 3;
            ASYNC16(&Vl[buf][r][c * 8], vp + (size_t)r * KV_ + kb + c * 8);
        }
    };

    // Resident Q^T B-operands: lane n holds query qbase+n, halves hi8+0..7 and hi8+16..23
    v16h bq[4];
    const _Float16* __restrict__ qrow =
        q + ((size_t)(bh * Q_ + qbase + l16)) * D_;
    #pragma unroll
    for (int c = 0; c < 4; ++c)
        bq[c] = __builtin_shufflevector(*(const v8h*)(qrow + c * 32 + hi8), *(const v8h*)(qrow + c * 32 + hi8 + 16), 0, 1, 2, 3, 4, 5, 6, 7, 8, 9, 10, 11, 12, 13, 14, 15);

    v8f acc[8] = {};                 // O^T accumulators: 8 d-tiles x 16 queries
    float m_run = -__builtin_inff();
    float l_run = 0.f;

    stage(0, 0);                     // prologue
    int cur = 0;

    #pragma unroll 1
    for (int it = 0; it < NCHUNK; ++it) {
        const int  nxt  = (cur + 1 == NBUF) ? 0 : cur + 1;
        const bool more = (it + 1) < NCHUNK;
        if (more) {
            stage(nxt, it + 1);        // prefetch next chunk into next buffer
            wait_async_le8();          // current chunk's 8 copies are done
        } else {
            wait_async_0();
        }
        __syncthreads();               // all waves' copies for chunk `it` done

        // ---- batch-load all K A-operands (16 x ds_load_b128)
        const _Float16* kr0 = &Kl[cur][l16][hi8];
        const _Float16* kr1 = &Kl[cur][16 + l16][hi8];
        V16 a[4][2];
        #pragma unroll
        for (int c = 0; c < 4; ++c) {
            a[c][0].h[0] = *(const v8h*)(kr0 + c * 32);
            a[c][0].h[1] = *(const v8h*)(kr0 + c * 32 + 16);
            a[c][1].h[0] = *(const v8h*)(kr1 + c * 32);
            a[c][1].h[1] = *(const v8h*)(kr1 + c * 32 + 16);
        }
        // pin: force all 16 loads resident -> one wait, then 8 wmma burst
        asm volatile("" :: "v"(a[0][0].v), "v"(a[0][1].v),
                           "v"(a[1][0].v), "v"(a[1][1].v),
                           "v"(a[2][0].v), "v"(a[2][1].v),
                           "v"(a[3][0].v), "v"(a[3][1].v));

        // ---- S^T = K_tile * Q^T  (two 16-key tiles, contraction d=128)
        v8f s0 = {}, s1 = {};
        #pragma unroll
        for (int c = 0; c < 4; ++c) {
            s0 = __builtin_amdgcn_wmma_f32_16x16x32_f16(
                     false, a[c][0].v, false, bq[c], (short)0, s0, false, false);
            s1 = __builtin_amdgcn_wmma_f32_16x16x32_f16(
                     false, a[c][1].v, false, bq[c], (short)0, s1, false, false);
        }

        // ---- issue V^T A-operand loads now; latency hides under softmax
        V16 av[8];
        #pragma unroll
        for (int dt = 0; dt < 8; ++dt) {
            const _Float16* vr = &Vl[cur][dt * 16 + l16][hi8];
            av[dt].h[0] = *(const v8h*)(vr);
            av[dt].h[1] = *(const v8h*)(vr + 16);
        }

        s0 = s0 * scale;
        s1 = s1 * scale;

        // ---- online softmax over this 32-key chunk (per query = lane&15)
        float cm = -__builtin_inff();
        #pragma unroll
        for (int r = 0; r < 8; ++r) {
            cm = fmaxf(cm, s0[r]);
            cm = fmaxf(cm, s1[r]);
        }
        cm = fmaxf(cm, __shfl_xor(cm, 16, 32));     // other 16 keys
        const float m_new = fmaxf(m_run, cm);
        const float corr  = (m_run == -__builtin_inff())
                            ? 0.f : __expf(m_run - m_new);

        float p0[8], p1[8], ssum = 0.f;
        #pragma unroll
        for (int r = 0; r < 8; ++r) {
            p0[r] = __expf(s0[r] - m_new);
            p1[r] = __expf(s1[r] - m_new);
            ssum += p0[r] + p1[r];
        }
        ssum += __shfl_xor(ssum, 16, 32);
        l_run = l_run * corr + ssum;
        m_run = m_new;
        #pragma unroll
        for (int t = 0; t < 8; ++t) acc[t] = acc[t] * corr;

        // ---- pack P (f32 C-layout) -> P^T B-operand (f16, no lane swap)
        int h0i[4], h1i[4];
        #pragma unroll
        for (int j = 0; j < 4; ++j) {
            PK pk;
            pk.f[0] = (_Float16)p0[2 * j];
            pk.f[1] = (_Float16)p0[2 * j + 1];
            h0i[j] = pk.i;
            pk.f[0] = (_Float16)p1[2 * j];
            pk.f[1] = (_Float16)p1[2 * j + 1];
            h1i[j] = pk.i;
        }
        V16 bp;
        #pragma unroll
        for (int j = 0; j < 4; ++j) {
            const int t0 = __shfl_xor(h0i[j], 16, 32);
            const int t1 = __shfl_xor(h1i[j], 16, 32);
            bp.i[j]     = h0i[j];               // keys  0..7  | 8..15
            bp.i[4 + j] = h1i[j];               // keys 16..23 | 24..31
        }

        // pin V^T operands here: wait lands after softmax, before PV burst
        asm volatile("" :: "v"(av[0].v), "v"(av[1].v),
                           "v"(av[2].v), "v"(av[3].v),
                           "v"(av[4].v), "v"(av[5].v),
                           "v"(av[6].v), "v"(av[7].v));

        // ---- O^T += V^T_tile * P^T  (contraction = 32 keys, 8 d-tiles)
        #pragma unroll
        for (int dt = 0; dt < 8; ++dt) {
            acc[dt] = __builtin_amdgcn_wmma_f32_16x16x32_f16(
                          false, av[dt].v, false, bp.v, (short)0,
                          acc[dt], false, false);
        }
        cur = nxt;   // triple buffering: no trailing barrier needed
    }

    // ---- epilogue: normalize, store out (fp16) + lse (fp32)
    const float inv = 1.0f / l_run;
    const float lse = m_run + __logf(l_run);
    _Float16* __restrict__ orow =
        out + ((size_t)(bh * Q_ + qbase + l16)) * D_ + hi8;
    #pragma unroll
    for (int dt = 0; dt < 8; ++dt) {
        v8h o;
        #pragma unroll
        for (int r = 0; r < 8; ++r)
            o[r] = (_Float16)(acc[dt][r] * inv);
        *(v8h*)(orow + dt * 16) = o;
    }
    if (lo)
        lse_out[bh * Q_ + qbase + lane] = lse;
}

// ------------------------------- launcher ----------------------------------
extern "C" void kernel_launch(void* const* d_in, const int* in_sizes, int n_in,
                              void* d_out, int out_size, void* d_ws, size_t ws_size,
                              hipStream_t stream)
{
    const _Float16* q  = (const _Float16*)d_in[0];
    const _Float16* kA = (const _Float16*)d_in[1];
    const _Float16* vA = (const _Float16*)d_in[2];
    const _Float16* kB = (const _Float16*)d_in[3];
    const _Float16* vB = (const _Float16*)d_in[4];

    _Float16* vtA = (_Float16*)d_ws;                        // [BH][D][KV] fp16
    _Float16* vtB = vtA + (size_t)BH_ * D_ * KV_;           // +32 MB

    _Float16* outp = (_Float16*)d_out;                      // [B,H,Q,D] fp16
    float*    lsep = (float*)(outp + (size_t)BH_ * Q_ * D_);// [B,H,Q]  fp32

    dim3 gt(KV_ / 16, D_ / 16, BH_ * 2);
    vtrans_kernel<<<gt, 256, 0, stream>>>(vA, vB, vtA, vtB);

    dim3 ga(Q_ / 64, BH_);
    mattn_kernel<<<ga, 128, 0, stream>>>(q, kA, kB, vtA, vtB, outp, lsep);
}
